// GPT2_7748121002576
// MI455X (gfx1250) — matrix operand order repaired and checked by reading, not compiled, not run on hardware
//
#include <hip/hip_runtime.h>
#include <hip/hip_bf16.h>
#include <math.h>

// ---------------------------------------------------------------------------
// GPT-2 forward on gfx1250 (MI455X).
//   * All dense GEMMs: v_wmma_f32_16x16x32_f16, f32 accumulate, wave tile
//     32(M) x 64(N), 8 WMMAs per K-step.
//   * Operands pre-converted to f16 (weights zero-padded to N%64==0) so the
//     GEMM hot loop is a single branch-free path (b128 A loads, per-element B gathers).
//   * Residual stream (x) and qkv stay f32; LN / attention / GELU outputs
//     are written f16 since they only feed GEMM-A.
// ---------------------------------------------------------------------------

#define GB  2
#define GS  1024
#define GD  768
#define GFF 3072
#define GH  12
#define GHD 64
#define GNL 4
#define GV  50257
#define GVP 50304              // GV rounded up to 64
#define GT  (GB * GS)          // 2048 token rows

typedef __attribute__((ext_vector_type(16))) _Float16 v16h;
typedef __attribute__((ext_vector_type(8)))  _Float16 v8h;
typedef __attribute__((ext_vector_type(8)))  float    v8f;

union AFrag {                  // v16h assembled from two aligned 16B loads
    v16h v;
    v8h  h8[2];
};

// ---------------------------------------------------------------------------
// f32 -> f16 conversion with optional column padding (zero fill).
//   src: [rows, N] f32   dst: [rows, Np] f16   grid = (ceil(Np/256), rows)
// ---------------------------------------------------------------------------
__global__ __launch_bounds__(256) void gpt2_cvt_f16_kernel(
    const float* __restrict__ src, _Float16* __restrict__ dst, int N, int Np)
{
    const int col = blockIdx.x * 256 + threadIdx.x;
    const int row = blockIdx.y;
    if (col < Np) {
        const float v = (col < N) ? src[(size_t)row * N + col] : 0.0f;
        dst[(size_t)row * Np + col] = (_Float16)v;
    }
}

// ---------------------------------------------------------------------------
// Embedding: x[t,d] = tok_embed[tokens[t], d] + pos_embed[t % S, d]
// ---------------------------------------------------------------------------
__global__ __launch_bounds__(256) void gpt2_embed_kernel(
    const int* __restrict__ tokens,
    const float* __restrict__ tok_embed,
    const float* __restrict__ pos_embed,
    float* __restrict__ x)
{
    const int trow = blockIdx.x;            // 0..2047
    const int s    = trow % GS;
    const int tok  = tokens[trow];
    const float* te = tok_embed + (size_t)tok * GD;
    const float* pe = pos_embed + (size_t)s   * GD;
    float* xp = x + (size_t)trow * GD;
    for (int d = threadIdx.x; d < GD; d += blockDim.x)
        xp[d] = te[d] + pe[d];
}

// ---------------------------------------------------------------------------
// LayerNorm over last dim (D=768), one block per token row; f16 output
// (feeds GEMM-A only).
// ---------------------------------------------------------------------------
__global__ __launch_bounds__(256) void gpt2_layernorm_kernel(
    const float* __restrict__ x,
    const float* __restrict__ gamma,
    const float* __restrict__ beta,
    _Float16* __restrict__ out)
{
    const int row = blockIdx.x;
    const int t   = threadIdx.x;
    __shared__ float s1[256];
    __shared__ float s2[256];

    const float* xp = x + (size_t)row * GD;
    float a = 0.f, b = 0.f;
    for (int d = t; d < GD; d += 256) {
        float v = xp[d];
        a += v;
        b += v * v;
    }
    s1[t] = a; s2[t] = b;
    __syncthreads();
    for (int off = 128; off > 0; off >>= 1) {
        if (t < off) { s1[t] += s1[t + off]; s2[t] += s2[t + off]; }
        __syncthreads();
    }
    const float mean = s1[0] * (1.0f / GD);
    const float var  = s2[0] * (1.0f / GD) - mean * mean;
    const float inv  = rsqrtf(var + 1e-5f);

    _Float16* op = out + (size_t)row * GD;
    for (int d = t; d < GD; d += 256)
        op[d] = (_Float16)((xp[d] - mean) * inv * gamma[d] + beta[d]);
}

// ---------------------------------------------------------------------------
// WMMA GEMM:  C[M,N] = act( A[M,K] * W[K,Np] + bias[N] (+ resid[M,N]) )
//   A: f16 [M,K];  W: f16 [K,Np] zero-padded (Np % 64 == 0, K % 32 == 0).
//   Output: f32 (Cf) or f16 (Ch) -- exactly one non-null.
//   block = 256 threads = 8 waves; wave tile = 32(M) x 64(N).
// Fragment layouts per CDNA5 ISA (CDNA5 ISA):
//   A (16x32 f16): lane L -> row M = L&15; elems 0..7 -> K = 8*(L>>4)+e,
//                  elems 8..15 -> K = 16 + 8*(L>>4) + (e-8).
//   B (32x16 f16): lane L -> N-col = L&15; elem e -> same K as A elem e.
//   C (16x16 f32): VGPR r -> row = r + 8*(L>>4); col = L&15.
// ---------------------------------------------------------------------------
__global__ __launch_bounds__(256) void gpt2_gemm_wmma_kernel(
    const _Float16* __restrict__ A,
    const _Float16* __restrict__ W,
    const float* __restrict__ bias,      // may be nullptr
    const float* __restrict__ resid,     // may be nullptr
    float* __restrict__ Cf,              // f32 out or nullptr
    _Float16* __restrict__ Ch,           // f16 out or nullptr
    int M, int N, int Np, int K, int act)
{
    const int lane  = threadIdx.x & 31;
    const int wave  = threadIdx.x >> 5;
    const int mbase = blockIdx.y * 256 + wave * 32;
    const int nbase = blockIdx.x * 64;

    const int half  = lane >> 4;         // 0 or 1
    const int mlane = lane & 15;

    const _Float16* ar0 = A + (size_t)(mbase + mlane)      * K + half * 8;
    const _Float16* ar1 = A + (size_t)(mbase + 16 + mlane) * K + half * 8;

    v8f acc[2][4] = {{v8f{}, v8f{}, v8f{}, v8f{}},
                     {v8f{}, v8f{}, v8f{}, v8f{}}};

    for (int kk = 0; kk < K; kk += 32) {
        // two A fragments: 4 aligned 16B loads total
        AFrag a0, a1;
        a0.h8[0] = *reinterpret_cast<const v8h*>(ar0 + kk);
        a0.h8[1] = *reinterpret_cast<const v8h*>(ar0 + kk + 16);
        a1.h8[0] = *reinterpret_cast<const v8h*>(ar1 + kk);
        a1.h8[1] = *reinterpret_cast<const v8h*>(ar1 + kk + 16);

        // 4 B fragments (column n = lane&15, k gathered per A order), 8 WMMAs; B reused for both M tiles
        const _Float16* wr = W + (size_t)(kk + half * 8) * Np + nbase + mlane;
        #pragma unroll
        for (int j = 0; j < 4; ++j) {
            v16h b;
            #pragma unroll
            for (int s = 0; s < 16; ++s)
                b[s] = wr[(size_t)(s < 8 ? s : s + 8) * Np + j * 16];
            acc[0][j] = __builtin_amdgcn_wmma_f32_16x16x32_f16(
                false, a0.v, false, b, (short)0, acc[0][j], false, false);
            acc[1][j] = __builtin_amdgcn_wmma_f32_16x16x32_f16(
                false, a1.v, false, b, (short)0, acc[1][j], false, false);
        }
    }

    // ---- Epilogue: bias / residual / GELU, scatter per C-fragment layout --
    #pragma unroll
    for (int i = 0; i < 2; ++i) {
        #pragma unroll
        for (int j = 0; j < 4; ++j) {
            #pragma unroll
            for (int r = 0; r < 8; ++r) {
                const int row = mbase + i * 16 + r + half * 8;
                const int col = nbase + j * 16 + mlane;
                if (col < N) {
                    float v = acc[i][j][r];
                    if (bias)  v += bias[col];
                    if (resid) v += resid[(size_t)row * N + col];
                    if (act == 1) {
                        const float u = v;
                        v = 0.5f * u *
                            (1.0f + tanhf(0.7978845608028654f *
                                          (u + 0.044715f * u * u * u)));
                    }
                    if (Cf) Cf[(size_t)row * N + col] = v;
                    else    Ch[(size_t)row * N + col] = (_Float16)v;
                }
            }
        }
    }
}

// ---------------------------------------------------------------------------
// Fused softmax attention (no causal mask, matching reference).
//   qkv: f32 [T, 3D] rows (Q | K | V, heads of 64 within each D block)
//   out: f16 [T, D]  (feeds proj GEMM-A)
//   grid = (S, H, B), block = 128 threads; each thread owns 8 keys.
// ---------------------------------------------------------------------------
__global__ __launch_bounds__(128) void gpt2_attention_kernel(
    const float* __restrict__ qkv,
    _Float16* __restrict__ out)
{
    const int q  = blockIdx.x;
    const int hh = blockIdx.y;
    const int b  = blockIdx.z;
    const int t  = threadIdx.x;          // 0..127

    __shared__ float qs[GHD];
    __shared__ float probs[GS];
    __shared__ float red[128];

    const size_t rowbase = (size_t)(b * GS);

    const float* qp = qkv + (rowbase + q) * (3 * GD) + hh * GHD;
    if (t < GHD) qs[t] = qp[t];
    __syncthreads();

    // scores for this thread's 8 keys
    float sc[8];
    float mx = -1e30f;
    #pragma unroll
    for (int i = 0; i < 8; ++i) {
        const int k = t + i * 128;
        const float* kp = qkv + (rowbase + k) * (3 * GD) + GD + hh * GHD;
        float s = 0.f;
        #pragma unroll 8
        for (int d = 0; d < GHD; ++d) s += qs[d] * kp[d];
        s *= 0.125f;                    // 1/sqrt(64)
        sc[i] = s;
        mx = fmaxf(mx, s);
    }

    // block max
    red[t] = mx; __syncthreads();
    for (int off = 64; off > 0; off >>= 1) {
        if (t < off) red[t] = fmaxf(red[t], red[t + off]);
        __syncthreads();
    }
    mx = red[0];
    __syncthreads();

    // exp + block sum
    float sum = 0.f;
    #pragma unroll
    for (int i = 0; i < 8; ++i) {
        const float e = expf(sc[i] - mx);
        probs[t + i * 128] = e;
        sum += e;
    }
    red[t] = sum; __syncthreads();
    for (int off = 64; off > 0; off >>= 1) {
        if (t < off) red[t] += red[t + off];
        __syncthreads();
    }
    const float inv = 1.0f / red[0];

    // o[d] = sum_k p[k] * V[k,d]   (threads 0..63 each own one d)
    if (t < GHD) {
        float acc = 0.f;
        for (int k = 0; k < GS; ++k)
            acc += probs[k] * qkv[(rowbase + k) * (3 * GD) + 2 * GD + hh * GHD + t];
        out[(rowbase + q) * GD + hh * GHD + t] = (_Float16)(acc * inv);
    }
}

// ---------------------------------------------------------------------------
// Host driver
// ---------------------------------------------------------------------------
static inline void launch_cvt(const float* src, _Float16* dst, int rows,
                              int N, int Np, hipStream_t stream) {
    gpt2_cvt_f16_kernel<<<dim3((Np + 255) / 256, rows), 256, 0, stream>>>(
        src, dst, N, Np);
}

extern "C" void kernel_launch(void* const* d_in, const int* in_sizes, int n_in,
                              void* d_out, int out_size, void* d_ws, size_t ws_size,
                              hipStream_t stream) {
    (void)in_sizes; (void)n_in; (void)out_size; (void)ws_size;

    const int*   tokens    = (const int*)  d_in[0];
    const float* tok_embed = (const float*)d_in[1];
    const float* pos_embed = (const float*)d_in[2];
    const float* ln1_g     = (const float*)d_in[3];
    const float* ln1_b     = (const float*)d_in[4];
    const float* qkv_w     = (const float*)d_in[5];
    const float* qkv_b     = (const float*)d_in[6];
    const float* proj_w    = (const float*)d_in[7];
    const float* proj_b    = (const float*)d_in[8];
    const float* ln2_g     = (const float*)d_in[9];
    const float* ln2_b     = (const float*)d_in[10];
    const float* ff1_w     = (const float*)d_in[11];
    const float* ff1_b     = (const float*)d_in[12];
    const float* ff2_w     = (const float*)d_in[13];
    const float* ff2_b     = (const float*)d_in[14];
    const float* lm_head_w = (const float*)d_in[15];
    float* logits = (float*)d_out;

    // workspace layout (all offsets 64B-aligned by construction):
    //   x    f32 [2048, 768]        6,291,456 B
    //   qkv  f32 [2048, 2304]      18,874,368 B
    //   a16a f16 [2048, 768]        3,145,728 B   (LN out / attn out / x16)
    //   a16b f16 [2048, 3072]      12,582,912 B   (FF1 out)
    //   w16  f16 [K, Np] scratch   77,266,944 B   (max: 768 x 50304)
    char* wsb = (char*)d_ws;
    float*    x    = (float*)wsb;
    float*    qkv  = (float*)(wsb + (size_t)GT * GD * 4);
    _Float16* a16a = (_Float16*)(wsb + (size_t)GT * GD * 4
                                     + (size_t)GT * 3 * GD * 4);
    _Float16* a16b = (_Float16*)((char*)a16a + (size_t)GT * GD * 2);
    _Float16* w16  = (_Float16*)((char*)a16b + (size_t)GT * GFF * 2);

    const dim3 blk256(256);
    const int  mgrid = GT / 256;         // wave tile 32 rows, 8 waves/block

    // token + position embedding
    gpt2_embed_kernel<<<GT, blk256, 0, stream>>>(tokens, tok_embed, pos_embed, x);

    for (int l = 0; l < GNL; ++l) {
        const size_t lD = (size_t)l * GD;

        // ---- attention sub-block ----
        gpt2_layernorm_kernel<<<GT, blk256, 0, stream>>>(
            x, ln1_g + lD, ln1_b + lD, a16a);

        launch_cvt(qkv_w + (size_t)l * GD * 3 * GD, w16, GD, 3 * GD, 3 * GD, stream);
        gpt2_gemm_wmma_kernel<<<dim3(3 * GD / 64, mgrid), blk256, 0, stream>>>(
            a16a, w16, qkv_b + (size_t)l * 3 * GD, nullptr,
            qkv, nullptr, GT, 3 * GD, 3 * GD, GD, 0);

        gpt2_attention_kernel<<<dim3(GS, GH, GB), dim3(128), 0, stream>>>(qkv, a16a);

        launch_cvt(proj_w + (size_t)l * GD * GD, w16, GD, GD, GD, stream);
        gpt2_gemm_wmma_kernel<<<dim3(GD / 64, mgrid), blk256, 0, stream>>>(
            a16a, w16, proj_b + lD, x,
            x, nullptr, GT, GD, GD, GD, 0);

        // ---- MLP sub-block ----
        gpt2_layernorm_kernel<<<GT, blk256, 0, stream>>>(
            x, ln2_g + lD, ln2_b + lD, a16a);

        launch_cvt(ff1_w + (size_t)l * GD * GFF, w16, GD, GFF, GFF, stream);
        gpt2_gemm_wmma_kernel<<<dim3(GFF / 64, mgrid), blk256, 0, stream>>>(
            a16a, w16, ff1_b + (size_t)l * GFF, nullptr,
            nullptr, a16b, GT, GFF, GFF, GD, 1);

        launch_cvt(ff2_w + (size_t)l * GFF * GD, w16, GFF, GD, GD, stream);
        gpt2_gemm_wmma_kernel<<<dim3(GD / 64, mgrid), blk256, 0, stream>>>(
            a16b, w16, ff2_b + lD, x,
            x, nullptr, GT, GD, GD, GFF, 0);
    }

    // ---- lm_head: logits = x @ lm_head_w  [2048 x 50257] ----
    launch_cvt(x, a16a, GT, GD, GD, stream);
    launch_cvt(lm_head_w, w16, GD, GV, GVP, stream);
    gpt2_gemm_wmma_kernel<<<dim3(GVP / 64, mgrid), blk256, 0, stream>>>(
        a16a, w16, nullptr, nullptr,
        logits, nullptr, GT, GV, GVP, GD, 0);
}
